// SelfAttentionBlock_8796093022424
// MI455X (gfx1250) — matrix operand order repaired and checked by reading, not compiled, not run on hardware
//
#include <hip/hip_runtime.h>
#include <math.h>

typedef _Float16 f16;
typedef __attribute__((ext_vector_type(16))) _Float16 v16h;
typedef __attribute__((ext_vector_type(8)))  float    v8f;

#define B_    8
#define N_    1024
#define C_    256
#define H_    8
#define D_    32
#define F_    682
#define FPAD_ 704
#define M_TOT (B_ * N_)   // 8192 rows

// -------------------- WMMA helpers --------------------

__device__ __forceinline__ v8f wmma32(v16h a, v16h b, v8f c) {
  // D = A(16x32 f16) * B(32x16 f16) + C(16x16 f32)
  return __builtin_amdgcn_wmma_f32_16x16x32_f16(false, a, false, b, (short)0, c,
                                                false, false);
}

// Position of element (M, kk) of a 16x32 A tile inside the 512-element
// swizzled blob, such that lane L loads v16h at offset L*16 and gets exactly
// the contiguous k order its B operands also use:
//   lane<16 : K in {0..15};  lane>=16 : K in {16..31}
//   element i -> K = i (+16 hi)
__device__ __forceinline__ int a_swz_pos(int M, int kk) {
  int hi  = (kk >> 4) & 1;
  int i   = kk & 15;
  return (hi * 16 + M) * 16 + i;
}

__device__ __forceinline__ float wave_sum32(float v) {
#pragma unroll
  for (int off = 16; off >= 1; off >>= 1) v += __shfl_xor(v, off, 32);
  return v;
}

// async global -> LDS copy of 16 bytes per lane (CDNA5 GLOBAL_LOAD_ASYNC_TO_LDS_B128)
__device__ __forceinline__ void async_ld_lds_b128(float* ldsp, const float* gsrc) {
  unsigned ldsoff =
      (unsigned)(size_t)(__attribute__((address_space(3))) float*)ldsp;
  asm volatile("global_load_async_to_lds_b128 %0, %1, off"
               :: "v"(ldsoff), "v"(gsrc) : "memory");
}
__device__ __forceinline__ void wait_asynccnt0() {
  asm volatile("s_wait_asynccnt 0x0" ::: "memory");
}

// -------------------- weight prep: f32 [K,N] -> f16 [N,Kpad] (transposed, zero-padded K)

__global__ void transpose_w_kernel(const float* __restrict__ src, f16* __restrict__ dst,
                                   int K, int Nn, int Kpad) {
  int idx = blockIdx.x * blockDim.x + threadIdx.x;
  int total = Nn * Kpad;
  if (idx >= total) return;
  int n = idx / Kpad, kk = idx % Kpad;
  float v = (kk < K) ? src[(size_t)kk * Nn + n] : 0.f;
  dst[idx] = (f16)v;
}

// -------------------- LayerNorm over C=256, f32 in -> swizzled f16 A-layout out --

__global__ __launch_bounds__(256)
void ln256_kernel(const float* __restrict__ x, const float* __restrict__ g,
                  const float* __restrict__ bb, f16* __restrict__ outswz) {
  int row = blockIdx.x, t = threadIdx.x;
  int lane = t & 31, wid = t >> 5;
  float v = x[(size_t)row * C_ + t];
  __shared__ float red[8];
  float ws = wave_sum32(v);
  if (lane == 0) red[wid] = ws;
  __syncthreads();
  float tot = 0.f;
#pragma unroll
  for (int i = 0; i < 8; ++i) tot += red[i];
  float mean = tot * (1.f / C_);
  float d = v - mean;
  __syncthreads();
  float ws2 = wave_sum32(d * d);
  if (lane == 0) red[wid] = ws2;
  __syncthreads();
  float tv = 0.f;
#pragma unroll
  for (int i = 0; i < 8; ++i) tv += red[i];
  float y = d * rsqrtf(tv * (1.f / C_) + 1e-3f) * g[t] + bb[t];
  size_t blob = ((size_t)(row >> 4) * (C_ / 32) + (t >> 5)) * 512;
  outswz[blob + a_swz_pos(row & 15, t & 31)] = (f16)y;
}

// -------------------- GeGLU combine + LayerNorm over F=682 -> swizzled f16 ------

__global__ __launch_bounds__(256)
void geglu_lnf_kernel(const float* __restrict__ wide, const float* __restrict__ gate,
                      const float* __restrict__ g, const float* __restrict__ bb,
                      f16* __restrict__ outswz) {
  int row = blockIdx.x, t = threadIdx.x;
  int lane = t & 31, wid = t >> 5;
  float vals[3];
  float s = 0.f;
#pragma unroll
  for (int i = 0; i < 3; ++i) {
    int c = t + i * 256;
    float f = 0.f;
    if (c < F_) {
      float w  = wide[(size_t)row * FPAD_ + c];
      float gt = gate[(size_t)row * FPAD_ + c];
      f = 0.5f * w * (1.f + erff(w * 0.70710678118654752f)) * gt;  // exact gelu * gate
    }
    vals[i] = f; s += f;
  }
  __shared__ float red[8];
  float ws = wave_sum32(s);
  if (lane == 0) red[wid] = ws;
  __syncthreads();
  float tot = 0.f;
#pragma unroll
  for (int i = 0; i < 8; ++i) tot += red[i];
  float mean = tot * (1.f / F_);
  __syncthreads();
  float sv = 0.f;
#pragma unroll
  for (int i = 0; i < 3; ++i) {
    int c = t + i * 256;
    if (c < F_) { float d = vals[i] - mean; sv += d * d; }
  }
  float ws2 = wave_sum32(sv);
  if (lane == 0) red[wid] = ws2;
  __syncthreads();
  float tv = 0.f;
#pragma unroll
  for (int i = 0; i < 8; ++i) tv += red[i];
  float rstd = rsqrtf(tv * (1.f / F_) + 1e-3f);
#pragma unroll
  for (int i = 0; i < 3; ++i) {
    int c = t + i * 256;
    if (c < FPAD_) {
      f16 o = (f16)0;
      if (c < F_) o = (f16)((vals[i] - mean) * rstd * g[c] + bb[c]);
      size_t blob = ((size_t)(row >> 4) * (FPAD_ / 32) + (c >> 5)) * 512;
      outswz[blob + a_swz_pos(row & 15, c & 31)] = o;   // zero K-padding included
    }
  }
}

// -------------------- generic WMMA GEMM: out = A[M,Kpad] @ BT[N,Kpad]^T ----------
// A is stored in swizzled A-operand blobs: one v16h load per 16x32 tile per lane.

constexpr int EPI_F32 = 0, EPI_QKV = 1, EPI_OUTPROJ = 2, EPI_DOWN = 3;

template <int EPI>
__global__ __launch_bounds__(256)
void gemm_wmma(const f16* __restrict__ Aswz, const f16* __restrict__ BT,
               int Nw, int Kpad, int ldo,
               const float* __restrict__ bias,
               float* __restrict__ outF,
               const float* __restrict__ resid,
               const float* __restrict__ scale,
               f16* __restrict__ qswz, f16* __restrict__ kP, f16* __restrict__ vTP) {
  constexpr bool NB = (EPI == EPI_F32);   // only wide/gate (Nw=682) need bounds
  const int lane = threadIdx.x & 31;
  const int wid  = threadIdx.x >> 5;
  const int hi   = (lane >> 4) & 1;
  const int ln   = lane & 15;
  const int rowbase = blockIdx.x * 128 + (wid & 3) * 32;  // M tiled 128/block, 32/wave
  const int colbase = blockIdx.y * 64 + (wid >> 2) * 32;  // N tiled  64/block, 32/wave
  const int nch = Kpad >> 5;
  const v16h vzero = {};
  v8f acc[2][2] = {};

  const f16* abase = Aswz + ((size_t)(rowbase >> 4) * nch) * 512 + lane * 16;
  for (int k0 = 0; k0 < Kpad; k0 += 32) {
    v16h a0 = *(const v16h*)(abase);
    v16h a1 = *(const v16h*)(abase + (size_t)nch * 512);
    abase += 512;
    int c0 = colbase + ln, c1 = c0 + 16;
    v16h b0, b1;
    if (NB) {
      int c0c = (c0 < Nw) ? c0 : 0;
      int c1c = (c1 < Nw) ? c1 : 0;
      v16h t0 = *(const v16h*)(BT + (size_t)c0c * Kpad + k0 + (hi ? 16 : 0));
      v16h t1 = *(const v16h*)(BT + (size_t)c1c * Kpad + k0 + (hi ? 16 : 0));
      b0 = (c0 < Nw) ? t0 : vzero;
      b1 = (c1 < Nw) ? t1 : vzero;
    } else {
      b0 = *(const v16h*)(BT + (size_t)c0 * Kpad + k0 + (hi ? 16 : 0));
      b1 = *(const v16h*)(BT + (size_t)c1 * Kpad + k0 + (hi ? 16 : 0));
    }
    acc[0][0] = wmma32(a0, b0, acc[0][0]);
    acc[0][1] = wmma32(a0, b1, acc[0][1]);
    acc[1][0] = wmma32(a1, b0, acc[1][0]);
    acc[1][1] = wmma32(a1, b1, acc[1][1]);
  }

#pragma unroll
  for (int mt = 0; mt < 2; ++mt)
#pragma unroll
    for (int nt = 0; nt < 2; ++nt)
#pragma unroll
      for (int r = 0; r < 8; ++r) {
        int row = rowbase + mt * 16 + r + hi * 8;   // C/D layout: M = r + 8*hi
        int col = colbase + nt * 16 + ln;           //            N = lane%16
        if (NB && col >= Nw) continue;
        float v = acc[mt][nt][r];
        if (EPI == EPI_F32) {
          outF[(size_t)row * ldo + col] = v;
        } else if (EPI == EPI_QKV) {
          v += bias[col];
          int seg = col >> 8;          // 0=q 1=k 2=v
          int c2  = col & 255;
          int h = c2 >> 5, d = c2 & 31;
          int b = row >> 10, n = row & 1023;
          size_t bh = (size_t)(b * H_ + h);
          if (seg == 0) {
            // q in swizzled A-operand blobs, pre-scaled by 1/sqrt(D)
            size_t blob = (bh * (N_ / 16) + (n >> 4)) * 512;
            qswz[blob + a_swz_pos(n & 15, d)] = (f16)(v * 0.17677669529663687f);
          } else if (seg == 1) {
            kP[(bh * N_ + n) * D_ + d] = (f16)v;
          } else {
            vTP[(bh * D_ + d) * N_ + n] = (f16)v;  // V transposed for AV B-operand
          }
        } else if (EPI == EPI_OUTPROJ) {
          v += bias[col];
          size_t idx = (size_t)row * C_ + col;
          outF[idx] = resid[idx] + scale[col] * v;   // x = inputs + gamma1*(attn@Wout+bout)
        } else {  // EPI_DOWN
          size_t idx = (size_t)row * C_ + col;
          outF[idx] = resid[idx] + scale[col] * v;   // out = x + gamma2*(f@W_down)
        }
      }
}

// -------------------- fused talking-heads attention ------------------------------
// One block = 8 waves = 8 heads, handling a 16-row query tile of one batch.
// Two-pass online softmax; head mixing (Wt1 pre-softmax, Wt2 post-softmax) via LDS.
// The interaction tile [16n x 32m x 8h] is copied global->LDS with CDNA5 async
// loads (no VGPR staging, tracked by ASYNCcnt) into a double buffer so the copy
// overlaps the WMMAs without racing the previous chunk's consumers.

__global__ __launch_bounds__(256)
void attn_kernel(const f16* __restrict__ qswz, const f16* __restrict__ k,
                 const f16* __restrict__ vT,
                 const float* __restrict__ inter, const unsigned char* __restrict__ mask,
                 const float* __restrict__ Wt1, const float* __restrict__ bt1,
                 const float* __restrict__ Wt2, const float* __restrict__ bt2,
                 f16* __restrict__ aoswz) {
  const int b  = blockIdx.x >> 6;          // N_/16 = 64 row tiles per batch
  const int n0 = (blockIdx.x & 63) * 16;
  const int t    = threadIdx.x;
  const int lane = t & 31;
  const int g    = t >> 5;                 // head index = wave id
  const int hi   = (lane >> 4) & 1;
  const int ln   = lane & 15;

  __shared__ float sS[H_][16][32];                         // score/prob tiles (16 KB)
  __shared__ __align__(16) float sInter[2][16 * 32 * H_];  // interaction dbl-buf (32 KB)
  __shared__ float sColAdd[N_];                            // 0.f / -1e9f (4 KB)

  for (int j = t; j < N_; j += 256)
    sColAdd[j] = mask[(size_t)b * N_ * N_ + j] ? 0.f : -1e9f;  // mask[b][0][j]==valid[j]

  float w1[H_], w2[H_];
#pragma unroll
  for (int h = 0; h < H_; ++h) { w1[h] = Wt1[h * H_ + g]; w2[h] = Wt2[h * H_ + g]; }
  const float b1g = bt1[g], b2g = bt2[g];

  const size_t bh = (size_t)(b * H_ + g);
  const v16h aQ = *(const v16h*)(qswz + (bh * (N_ / 16) + (n0 >> 4)) * 512 + lane * 16);

  __syncthreads();

  // row mask terms (0 / -1e9); exact Keras mask: term = min(rowAdd, colAdd)
  float rAdd[8];
#pragma unroll
  for (int r = 0; r < 8; ++r) rAdd[r] = sColAdd[n0 + r + hi * 8];

  float rmax[8], rsum[8];
#pragma unroll
  for (int r = 0; r < 8; ++r) { rmax[r] = -3.0e38f; rsum[r] = 0.f; }

  // ---------------- pass 1: row max / sum-exp (online) ----------------
  for (int mc = 0; mc < N_ / 32; ++mc) {
    const int m0 = mc * 32;
    float* buf = sInter[mc & 1];
    // async global->LDS copy of the 16 KB interaction tile (coalesced b128/lane)
#pragma unroll
    for (int i = 0; i < 4; ++i) {
      int idx = i * 256 + t;
      const float* gsrc =
          inter + (((size_t)b * N_ + n0 + (idx >> 6)) * N_ + m0) * H_ + (idx & 63) * 4;
      async_ld_lds_b128(&buf[idx * 4], gsrc);
      if (i == 0 && mc + 1 < N_ / 32) __builtin_prefetch(gsrc + 32 * H_, 0, 0);
    }
    v8f s0 = {}, s1 = {};
    v16h bK0 = *(const v16h*)(k + (bh * N_ + m0 + ln) * D_ + (hi ? 16 : 0));
    v16h bK1 = *(const v16h*)(k + (bh * N_ + m0 + 16 + ln) * D_ + (hi ? 16 : 0));
    s0 = wmma32(aQ, bK0, s0);
    s1 = wmma32(aQ, bK1, s1);
    __syncthreads();
#pragma unroll
    for (int r = 0; r < 8; ++r) {
      int M = r + hi * 8;
      sS[g][M][ln]      = s0[r];
      sS[g][M][16 + ln] = s1[r];
    }
    wait_asynccnt0();
    __syncthreads();
    float cAdd0 = sColAdd[m0 + ln];
    float cAdd1 = sColAdd[m0 + 16 + ln];
#pragma unroll
    for (int r = 0; r < 8; ++r) {
      int M = r + hi * 8;
#pragma unroll
      for (int mt = 0; mt < 2; ++mt) {
        int mm = mt * 16 + ln;
        float v = b1g;
#pragma unroll
        for (int h = 0; h < H_; ++h) v += sS[h][M][mm] * w1[h];   // pre-softmax head mix
        v += buf[(M * 32 + mm) * H_ + g];
        v += fminf(rAdd[r], mt ? cAdd1 : cAdd0);                  // branchless exact mask
        if (v > rmax[r]) { rsum[r] *= __expf(rmax[r] - v); rmax[r] = v; }
        rsum[r] += __expf(v - rmax[r]);
      }
    }
  }
  // combine row stats across the 16 lanes of each half-wave
#pragma unroll
  for (int r = 0; r < 8; ++r) {
#pragma unroll
    for (int off = 8; off >= 1; off >>= 1) {
      float om = __shfl_xor(rmax[r], off, 32);
      float os = __shfl_xor(rsum[r], off, 32);
      float nm = fmaxf(rmax[r], om);
      rsum[r] = rsum[r] * __expf(rmax[r] - nm) + os * __expf(om - nm);
      rmax[r] = nm;
    }
    rsum[r] = 1.f / rsum[r];
  }
  __syncthreads();   // pass-1 consumers done before pass-2 overwrites buffers

  // ---------------- pass 2: probabilities -> Wt2 mix -> AV ----------------
  v8f O0 = {}, O1 = {};
  for (int mc = 0; mc < N_ / 32; ++mc) {
    const int m0 = mc * 32;
    float* buf = sInter[mc & 1];
#pragma unroll
    for (int i = 0; i < 4; ++i) {
      int idx = i * 256 + t;
      const float* gsrc =
          inter + (((size_t)b * N_ + n0 + (idx >> 6)) * N_ + m0) * H_ + (idx & 63) * 4;
      async_ld_lds_b128(&buf[idx * 4], gsrc);
    }
    v8f s0 = {}, s1 = {};
    v16h bK0 = *(const v16h*)(k + (bh * N_ + m0 + ln) * D_ + (hi ? 16 : 0));
    v16h bK1 = *(const v16h*)(k + (bh * N_ + m0 + 16 + ln) * D_ + (hi ? 16 : 0));
    s0 = wmma32(aQ, bK0, s0);
    s1 = wmma32(aQ, bK1, s1);
    __syncthreads();
#pragma unroll
    for (int r = 0; r < 8; ++r) {
      int M = r + hi * 8;
      sS[g][M][ln]      = s0[r];
      sS[g][M][16 + ln] = s1[r];
    }
    wait_asynccnt0();
    __syncthreads();
    float cAdd0 = sColAdd[m0 + ln];
    float cAdd1 = sColAdd[m0 + 16 + ln];
    float pv[2][8];
#pragma unroll
    for (int r = 0; r < 8; ++r) {
      int M = r + hi * 8;
#pragma unroll
      for (int mt = 0; mt < 2; ++mt) {
        int mm = mt * 16 + ln;
        float v = b1g;
#pragma unroll
        for (int h = 0; h < H_; ++h) v += sS[h][M][mm] * w1[h];
        v += buf[(M * 32 + mm) * H_ + g];
        v += fminf(rAdd[r], mt ? cAdd1 : cAdd0);
        pv[mt][r] = __expf(v - rmax[r]) * rsum[r];   // normalized attention
      }
    }
    __syncthreads();
#pragma unroll
    for (int r = 0; r < 8; ++r) {
      int M = r + hi * 8;
      sS[g][M][ln]      = pv[0][r];
      sS[g][M][16 + ln] = pv[1][r];
    }
    __syncthreads();
    // build post-mix probabilities directly in A-operand layout (+bt2 bias, exact:
    // all m columns are swept, so bt2*colsum(V) is accumulated by the WMMAs)
    v16h pa;
#pragma unroll
    for (int s = 0; s < 8; ++s) {
      int kk = 2 * s + hi * 16;
#pragma unroll
      for (int c = 0; c < 2; ++c) {
        float v = b2g;
#pragma unroll
        for (int h = 0; h < H_; ++h) v += sS[h][ln][kk + c] * w2[h];
        pa[2 * s + c] = (f16)v;
      }
    }
    v16h bV0 = *(const v16h*)(vT + (bh * D_ + ln) * N_ + m0 + (hi ? 16 : 0));
    v16h bV1 = *(const v16h*)(vT + (bh * D_ + 16 + ln) * N_ + m0 + (hi ? 16 : 0));
    O0 = wmma32(pa, bV0, O0);
    O1 = wmma32(pa, bV1, O1);
  }

  // write attention output in swizzled A-layout for the out-proj GEMM:
  // global row = b*N + n0 + M, col = g*32 + {ln, 16+ln}  ->  chunk == g
#pragma unroll
  for (int r = 0; r < 8; ++r) {
    int M = r + hi * 8;
    size_t tile = (size_t)((b * N_ + n0) >> 4);
    size_t blob = (tile * (C_ / 32) + g) * 512;
    aoswz[blob + a_swz_pos(M, ln)]      = (f16)O0[r];
    aoswz[blob + a_swz_pos(M, 16 + ln)] = (f16)O1[r];
  }
}

// -------------------- host launcher --------------------

extern "C" void kernel_launch(void* const* d_in, const int* in_sizes, int n_in,
                              void* d_out, int out_size, void* d_ws, size_t ws_size,
                              hipStream_t stream) {
  const float* inputs = (const float*)d_in[0];
  const unsigned char* mask = (const unsigned char*)d_in[1];
  const float* inter  = (const float*)d_in[2];
  const float* Wqkv   = (const float*)d_in[3];
  const float* bqkv   = (const float*)d_in[4];
  const float* Wt1    = (const float*)d_in[5];
  const float* bt1    = (const float*)d_in[6];
  const float* Wt2    = (const float*)d_in[7];
  const float* bt2    = (const float*)d_in[8];
  const float* Wout   = (const float*)d_in[9];
  const float* bout   = (const float*)d_in[10];
  const float* ln1_g  = (const float*)d_in[11];
  const float* ln1_b  = (const float*)d_in[12];
  const float* ln2_g  = (const float*)d_in[13];
  const float* ln2_b  = (const float*)d_in[14];
  const float* gamma1 = (const float*)d_in[15];
  const float* gamma2 = (const float*)d_in[16];
  const float* W_wide = (const float*)d_in[17];
  const float* W_gate = (const float*)d_in[18];
  const float* W_down = (const float*)d_in[19];
  const float* lnf_g  = (const float*)d_in[20];
  const float* lnf_b  = (const float*)d_in[21];
  float* out = (float*)d_out;

  char* ws = (char*)d_ws;
  size_t off = 0;
  auto alloc = [&](size_t bytes) {
    char* p = ws + off;
    off += (bytes + 255) & ~(size_t)255;
    return p;
  };
  f16*   hbuf    = (f16*)alloc((size_t)M_TOT * C_ * sizeof(f16));        // swizzled
  f16*   qbuf    = (f16*)alloc((size_t)B_ * H_ * N_ * D_ * sizeof(f16)); // swizzled
  f16*   kbuf    = (f16*)alloc((size_t)B_ * H_ * N_ * D_ * sizeof(f16));
  f16*   vTbuf   = (f16*)alloc((size_t)B_ * H_ * N_ * D_ * sizeof(f16));
  f16*   attnout = (f16*)alloc((size_t)M_TOT * C_ * sizeof(f16));        // swizzled
  float* xbuf    = (float*)alloc((size_t)M_TOT * C_ * sizeof(float));
  f16*   h2buf   = (f16*)alloc((size_t)M_TOT * C_ * sizeof(f16));        // swizzled
  float* widebuf = (float*)alloc((size_t)M_TOT * FPAD_ * sizeof(float));
  float* gatebuf = (float*)alloc((size_t)M_TOT * FPAD_ * sizeof(float));
  f16*   fhbuf   = (f16*)alloc((size_t)M_TOT * FPAD_ * sizeof(f16));     // swizzled
  f16*   WqkvT   = (f16*)alloc((size_t)768 * 256 * sizeof(f16));
  f16*   WoutT   = (f16*)alloc((size_t)256 * 256 * sizeof(f16));
  f16*   WwideT  = (f16*)alloc((size_t)F_ * 256 * sizeof(f16));
  f16*   WgateT  = (f16*)alloc((size_t)F_ * 256 * sizeof(f16));
  f16*   WdownT  = (f16*)alloc((size_t)256 * FPAD_ * sizeof(f16));

  auto blocks = [](int total) { return dim3((unsigned)((total + 255) / 256)); };

  // 1) weight prep (transpose + f32->f16, K zero-padded)
  transpose_w_kernel<<<blocks(768 * 256), 256, 0, stream>>>(Wqkv,   WqkvT,  256, 768, 256);
  transpose_w_kernel<<<blocks(256 * 256), 256, 0, stream>>>(Wout,   WoutT,  256, 256, 256);
  transpose_w_kernel<<<blocks(F_ * 256),  256, 0, stream>>>(W_wide, WwideT, 256, F_,  256);
  transpose_w_kernel<<<blocks(F_ * 256),  256, 0, stream>>>(W_gate, WgateT, 256, F_,  256);
  transpose_w_kernel<<<blocks(256 * FPAD_), 256, 0, stream>>>(W_down, WdownT, F_, 256, FPAD_);

  // 2) pre-norm -> swizzled f16
  ln256_kernel<<<dim3(M_TOT), 256, 0, stream>>>(inputs, ln1_g, ln1_b, hbuf);

  // 3) QKV projection (q swizzled + scaled by 1/sqrt(D); k row-major; v transposed)
  gemm_wmma<EPI_QKV><<<dim3(M_TOT / 128, 12), 256, 0, stream>>>(
      hbuf, WqkvT, 768, 256, 0, bqkv, nullptr, nullptr, nullptr, qbuf, kbuf, vTbuf);

  // 4) fused talking-heads attention -> swizzled attn output
  attn_kernel<<<dim3(B_ * (N_ / 16)), 256, 0, stream>>>(
      qbuf, kbuf, vTbuf, inter, mask, Wt1, bt1, Wt2, bt2, attnout);

  // 5) output projection + LayerScale residual: x = inputs + gamma1*(attn@Wout+bout)
  gemm_wmma<EPI_OUTPROJ><<<dim3(M_TOT / 128, 4), 256, 0, stream>>>(
      attnout, WoutT, 256, 256, 0, bout, xbuf, inputs, gamma1, nullptr, nullptr, nullptr);

  // 6) FFN pre-norm -> swizzled f16
  ln256_kernel<<<dim3(M_TOT), 256, 0, stream>>>(xbuf, ln2_g, ln2_b, h2buf);

  // 7) wide & gate projections
  gemm_wmma<EPI_F32><<<dim3(M_TOT / 128, (F_ + 63) / 64), 256, 0, stream>>>(
      h2buf, WwideT, F_, 256, FPAD_, nullptr, widebuf, nullptr, nullptr, nullptr, nullptr, nullptr);
  gemm_wmma<EPI_F32><<<dim3(M_TOT / 128, (F_ + 63) / 64), 256, 0, stream>>>(
      h2buf, WgateT, F_, 256, FPAD_, nullptr, gatebuf, nullptr, nullptr, nullptr, nullptr, nullptr);

  // 8) gelu(wide)*gate + inner LayerNorm -> swizzled f16 (K padded to 704, zeros)
  geglu_lnf_kernel<<<dim3(M_TOT), 256, 0, stream>>>(widebuf, gatebuf, lnf_g, lnf_b, fhbuf);

  // 9) down projection + LayerScale residual -> final output
  gemm_wmma<EPI_DOWN><<<dim3(M_TOT / 128, 4), 256, 0, stream>>>(
      fhbuf, WdownT, 256, FPAD_, 0, nullptr, out, xbuf, gamma2, nullptr, nullptr, nullptr);

  (void)in_sizes; (void)n_in; (void)out_size; (void)ws_size;
}
